// LSTMCore_two_layer_gate_24008867184791
// MI455X (gfx1250) — compile-verified, matrix operand order checked by reading, not run on hardware
//
#include <hip/hip_runtime.h>
#include <hip/hip_bf16.h>
#include <math.h>

// Problem constants (match reference)
constexpr int CB = 1024;  // batch
constexpr int CL = 40;    // frames
constexpr int CE = 1024;  // embed
constexpr int CR = 1024;  // rnn size
constexpr int CA = 512;   // attn size

typedef __attribute__((ext_vector_type(16))) __bf16 v16bf;
typedef __attribute__((ext_vector_type(8)))  float  v8f;
typedef __attribute__((ext_vector_type(4)))  unsigned tdm_g0_t;
typedef __attribute__((ext_vector_type(8)))  int      tdm_g1_t;
typedef __attribute__((ext_vector_type(4)))  int      tdm_g2_t;

union FragBF { v16bf v; unsigned u[8]; };

#if defined(__has_builtin)
#if __has_builtin(__builtin_amdgcn_tensor_load_to_lds) && __has_builtin(__builtin_amdgcn_s_wait_tensorcnt)
#define USE_TDM 1
#endif
#endif
#ifndef USE_TDM
#define USE_TDM 0
#endif

__device__ __forceinline__ __bf16 f2bf(float f) {
  unsigned u = __builtin_bit_cast(unsigned, f);
  unsigned r = u + 0x7FFFu + ((u >> 16) & 1u);  // round-to-nearest-even
  unsigned short h = (unsigned short)(r >> 16);
  return __builtin_bit_cast(__bf16, h);
}

// ---------------------------------------------------------------------------
// f32 -> bf16 conversion
// ---------------------------------------------------------------------------
__global__ __launch_bounds__(256) void cvt_bf16_kernel(
    const float* __restrict__ s, __bf16* __restrict__ d, size_t n) {
  size_t i = (size_t)blockIdx.x * blockDim.x + threadIdx.x;
  if (i < n) d[i] = f2bf(s[i]);
}

// Build [h1l | h2l] concat (bf16) plus individual bf16 copies
__global__ __launch_bounds__(256) void concat_h_kernel(
    const float* __restrict__ h1, const float* __restrict__ h2,
    __bf16* __restrict__ h12, __bf16* __restrict__ h1b, __bf16* __restrict__ h2b) {
  const size_t i = (size_t)blockIdx.x * blockDim.x + threadIdx.x;  // < CB*CR
  const int b = (int)(i >> 10);
  const int r = (int)(i & (CR - 1));
  const float a = h1[i];
  const float c = h2[i];
  h12[(size_t)b * (2 * CR) + r]      = f2bf(a);
  h12[(size_t)b * (2 * CR) + CR + r] = f2bf(c);
  h1b[i] = f2bf(a);
  h2b[i] = f2bf(c);
}

// ---------------------------------------------------------------------------
// bf16 WMMA GEMM:  C[M,N](f32) (=|+=) A[M,K](bf16,row) * B[K,N](bf16,row) + bias
// BM=128 BN=64 BK=32, 256 threads (8 wave32 waves, 4x2 wave grid, 32x32/wave)
// A tile is fetched by the Tensor Data Mover (TDM) directly into padded LDS;
// B tile is register-staged and transposed into LDS.
// Requires M%128==0, N%64==0, K%32==0 (true for all launches here).
// ---------------------------------------------------------------------------
constexpr int BM = 128, BN = 64, BK = 32;
constexpr int ASTR = 40;  // padded LDS stride (bf16 elems): 64B row + 16B pad = 80B
constexpr int BSTR = 40;

__global__ __launch_bounds__(256) void gemm_bf16_wmma(
    const __bf16* __restrict__ A, const __bf16* __restrict__ Bmat,
    float* __restrict__ C, const float* __restrict__ bias,
    int M, int N, int K, int accum) {
  __shared__ __attribute__((aligned(16))) __bf16 As[2][BM * ASTR];
  __shared__ __attribute__((aligned(16))) __bf16 Bs[2][BN * BSTR];  // transposed [n][k]

  const int tid   = threadIdx.x;
  const int wave  = tid >> 5;
  const int lane  = tid & 31;
  const int waveM = wave >> 1;     // 0..3 -> 32-row strip
  const int waveN = wave & 1;      // 0..1 -> 32-col strip
  const int lhalf = lane >> 4;     // 0/1
  const int l15   = lane & 15;
  const int base_k = lhalf * 8;    // per ISA 16-bit A/B fragment layout

  const int rowBase = blockIdx.y * BM;
  const int colBase = blockIdx.x * BN;

  v8f acc[2][2];
#pragma unroll
  for (int i = 0; i < 2; ++i)
#pragma unroll
    for (int j = 0; j < 2; ++j)
#pragma unroll
      for (int v = 0; v < 8; ++v) acc[i][j][v] = 0.0f;

  // ---- B tile staging: 32x64, one 8-wide bf16 chunk per thread ----
  uint4 rb;
  const int br = tid >> 3, bc = (tid & 7) * 8;
  auto loadGlobalB = [&](int kt) {
    rb = *(const uint4*)(Bmat + (size_t)(kt + br) * N + colBase + bc);
  };
  auto storeLDSB = [&](int buf) {
    auto st = [&](int j, unsigned u) {
      Bs[buf][(bc + 2 * j) * BSTR + br]     = __builtin_bit_cast(__bf16, (unsigned short)(u & 0xFFFFu));
      Bs[buf][(bc + 2 * j + 1) * BSTR + br] = __builtin_bit_cast(__bf16, (unsigned short)(u >> 16));
    };
    st(0, rb.x); st(1, rb.y); st(2, rb.z); st(3, rb.w);
  };

#if USE_TDM
  // ---- A tile via Tensor Data Mover: 2-D tile (32 x 128 rows), data_size=2B,
  // LDS padding: 16 DWORDs (64B row) then +4 DWORDs (16B) => 80B row = ASTR.
  auto issueTensorA = [&](int kt, int buf) {
    if (wave != 0) return;
    const unsigned lds = (unsigned)(unsigned long long)(const void*)&As[buf][0];
    const unsigned long long ga =
        (unsigned long long)(const void*)(A + (size_t)rowBase * K + kt);
    tdm_g0_t g0;
    g0.x = 1u;                                   // count=1 (valid), user mode
    g0.y = lds;                                  // lds_addr
    g0.z = (unsigned)(ga & 0xFFFFFFFFu);         // global_addr[31:0]
    g0.w = (unsigned)((ga >> 32) & 0x01FFFFFFu)  // global_addr[56:32]
         | (2u << 30);                           // type=2 ("image")
    tdm_g1_t g1;
    g1[0] = (1 << 16)                            // data_size: 2 bytes
          | (1 << 20)                            // pad_enable
          | (3 << 22)                            // pad_interval: 16 DWORDs (64B)
          | (3 << 25);                           // pad_amount: 4 DWORDs (16B)
    g1[1] = (int)(((unsigned)K & 0xFFFFu) << 16);            // tensor_dim0 lo16
    g1[2] = (int)(((unsigned)K >> 16) | (((unsigned)M & 0xFFFFu) << 16)); // td0 hi16 | td1 lo16
    g1[3] = (int)(((unsigned)M >> 16) | ((unsigned)BK << 16));            // td1 hi16 | tile_dim0=32
    g1[4] = BM;                                  // tile_dim1=128, tile_dim2=0
    g1[5] = K;                                   // tensor_dim0_stride[31:0]
    g1[6] = 0;                                   // stride hi | dim1_stride lo
    g1[7] = 0;
    tdm_g2_t gz4;
    gz4[0] = 0; gz4[1] = 0; gz4[2] = 0; gz4[3] = 0;  // 2-D tensor: groups 2/3 unused
    tdm_g1_t gz8;
#pragma unroll
    for (int i = 0; i < 8; ++i) gz8[i] = 0;
    // 6-arg toolchain form: (g0 v4u, g1 v8i, g2 v4i, g3 v4i, v8i, imm cpol)
    __builtin_amdgcn_tensor_load_to_lds(g0, g1, gz4, gz4, gz8, 0);
  };
#else
  // ---- Fallback: register-staged A tile (128x32 = 512 chunks of 8 bf16) ----
  uint4 ra0, ra1;
  const int ar0 = tid >> 2,         ac0 = (tid & 3) * 8;
  const int ar1 = (tid + 256) >> 2, ac1 = ((tid + 256) & 3) * 8;
  auto loadGlobalA = [&](int kt) {
    ra0 = *(const uint4*)(A + (size_t)(rowBase + ar0) * K + kt + ac0);
    ra1 = *(const uint4*)(A + (size_t)(rowBase + ar1) * K + kt + ac1);
  };
  auto storeLDSA = [&](int buf) {
    *(uint4*)&As[buf][ar0 * ASTR + ac0] = ra0;
    *(uint4*)&As[buf][ar1 * ASTR + ac1] = ra1;
  };
#endif

  // 16-bit fragment (16x32): lane l15 = row/col, base_k from lane half,
  // VGPR v holds K pair starting at base_k + (v<4 ? 2v : 8+2v)
  auto loadFragA = [&](int buf, int mt) {
    FragBF f;
    const __bf16* base = &As[buf][(waveM * 32 + mt * 16 + l15) * ASTR];
#pragma unroll
    for (int v = 0; v < 8; ++v) {
      const int k0 = base_k + ((v < 4) ? (2 * v) : (8 + 2 * v));
      f.u[v] = *(const unsigned*)(base + k0);
    }
    return f.v;
  };
  auto loadFragB = [&](int buf, int nt) {
    FragBF f;
    const __bf16* base = &Bs[buf][(waveN * 32 + nt * 16 + l15) * BSTR];
#pragma unroll
    for (int v = 0; v < 8; ++v) {
      const int k0 = base_k + ((v < 4) ? (2 * v) : (8 + 2 * v));
      f.u[v] = *(const unsigned*)(base + k0);
    }
    return f.v;
  };

  const int nk = K / BK;
#if USE_TDM
  issueTensorA(0, 0);
#else
  loadGlobalA(0);
#endif
  loadGlobalB(0);
#if !USE_TDM
  storeLDSA(0);
#endif
  storeLDSB(0);
#if USE_TDM
  if (wave == 0) __builtin_amdgcn_s_wait_tensorcnt(0);
#endif
  __syncthreads();

  for (int kt = 0; kt < nk; ++kt) {
    const int cur = kt & 1;
    if (kt + 1 < nk) {
#if USE_TDM
      issueTensorA((kt + 1) * BK, cur ^ 1);
#else
      loadGlobalA((kt + 1) * BK);
#endif
      loadGlobalB((kt + 1) * BK);
      if (kt + 2 < nk)
        __builtin_prefetch(Bmat + (size_t)((kt + 2) * BK + br) * N + colBase + bc, 0, 1);
    }
    v16bf a0 = loadFragA(cur, 0);
    v16bf a1 = loadFragA(cur, 1);
    v16bf b0 = loadFragB(cur, 0);
    v16bf b1 = loadFragB(cur, 1);
    acc[0][0] = __builtin_amdgcn_wmma_f32_16x16x32_bf16(false, a0, false, b0, (short)0, acc[0][0], false, false);
    acc[0][1] = __builtin_amdgcn_wmma_f32_16x16x32_bf16(false, a0, false, b1, (short)0, acc[0][1], false, false);
    acc[1][0] = __builtin_amdgcn_wmma_f32_16x16x32_bf16(false, a1, false, b0, (short)0, acc[1][0], false, false);
    acc[1][1] = __builtin_amdgcn_wmma_f32_16x16x32_bf16(false, a1, false, b1, (short)0, acc[1][1], false, false);
    if (kt + 1 < nk) {
#if !USE_TDM
      storeLDSA(cur ^ 1);
#endif
      storeLDSB(cur ^ 1);
#if USE_TDM
      if (wave == 0) __builtin_amdgcn_s_wait_tensorcnt(0);
#endif
    }
    __syncthreads();
  }

  // Epilogue: 16x16 f32 D layout — lane n = l15, VGPR v -> row v + 8*lhalf.
  // accum is wave-uniform: keep the branch outside the store loops.
  if (accum) {
#pragma unroll
    for (int tm = 0; tm < 2; ++tm)
#pragma unroll
      for (int tn = 0; tn < 2; ++tn) {
        const int col = colBase + waveN * 32 + tn * 16 + l15;
        const float bv = bias[col];
#pragma unroll
        for (int v = 0; v < 8; ++v) {
          const int row = rowBase + waveM * 32 + tm * 16 + lhalf * 8 + v;
          const size_t idx = (size_t)row * N + col;
          C[idx] += acc[tm][tn][v] + bv;
        }
      }
  } else {
#pragma unroll
    for (int tm = 0; tm < 2; ++tm)
#pragma unroll
      for (int tn = 0; tn < 2; ++tn) {
        const int col = colBase + waveN * 32 + tn * 16 + l15;
        const float bv = bias[col];
#pragma unroll
        for (int v = 0; v < 8; ++v) {
          const int row = rowBase + waveM * 32 + tm * 16 + lhalf * 8 + v;
          const size_t idx = (size_t)row * N + col;
          C[idx] = acc[tm][tn][v] + bv;
        }
      }
  }
}

// ---------------------------------------------------------------------------
// Attention: e[b,l] = tanh(hq[b,:] + vw[b,l,:]) . Wa2w + ba2w ; softmax over L
// One block (256 thr) per batch row. hq/vw already contain their biases.
// ---------------------------------------------------------------------------
__global__ __launch_bounds__(256) void attn_score_softmax(
    const float* __restrict__ hq, const float* __restrict__ vw,
    const float* __restrict__ wa2w, const float* __restrict__ ba2w,
    float* __restrict__ wout) {
  __shared__ float red[8];
  __shared__ float ebuf[CL];
  const int b = blockIdx.x, tid = threadIdx.x;
  const int lane = tid & 31, wv = tid >> 5;
  const float hq0 = hq[(size_t)b * CA + tid];
  const float hq1 = hq[(size_t)b * CA + 256 + tid];
  const float w0 = wa2w[tid], w1 = wa2w[256 + tid];
  for (int l = 0; l < CL; ++l) {
    const float* vp = vw + ((size_t)b * CL + l) * CA;
    float s = tanhf(hq0 + vp[tid]) * w0 + tanhf(hq1 + vp[256 + tid]) * w1;
#pragma unroll
    for (int o = 16; o > 0; o >>= 1) s += __shfl_xor(s, o, 32);
    if (lane == 0) red[wv] = s;
    __syncthreads();
    if (tid == 0) {
      float t = 0.f;
#pragma unroll
      for (int i = 0; i < 8; ++i) t += red[i];
      ebuf[l] = t + ba2w[0];
    }
    __syncthreads();
  }
  if (tid == 0) {
    float mx = ebuf[0];
    for (int l = 1; l < CL; ++l) mx = fmaxf(mx, ebuf[l]);
    float sum = 0.f;
    for (int l = 0; l < CL; ++l) { float v = expf(ebuf[l] - mx); ebuf[l] = v; sum += v; }
    const float inv = 1.0f / sum;
    for (int l = 0; l < CL; ++l) wout[(size_t)b * CL + l] = ebuf[l] * inv;
  }
}

// af[b,r] = sum_l w[b,l] * V[b,l,r]   (also emit bf16 copy)
__global__ __launch_bounds__(256) void attn_apply(
    const float* __restrict__ V, const float* __restrict__ w,
    float* __restrict__ af32, __bf16* __restrict__ afbf) {
  const int b = blockIdx.x, tid = threadIdx.x;
  float a0 = 0.f, a1 = 0.f, a2 = 0.f, a3 = 0.f;
  for (int l = 0; l < CL; ++l) {
    const float wl = w[(size_t)b * CL + l];
    const float* vp = V + ((size_t)b * CL + l) * CR;
    a0 += wl * vp[tid];
    a1 += wl * vp[tid + 256];
    a2 += wl * vp[tid + 512];
    a3 += wl * vp[tid + 768];
  }
  const size_t base = (size_t)b * CR;
  af32[base + tid]       = a0;  afbf[base + tid]       = f2bf(a0);
  af32[base + tid + 256] = a1;  afbf[base + tid + 256] = f2bf(a1);
  af32[base + tid + 512] = a2;  afbf[base + tid + 512] = f2bf(a2);
  af32[base + tid + 768] = a3;  afbf[base + tid + 768] = f2bf(a3);
}

// gated_pos = relu(g)*pos + pos  -> bf16
__global__ __launch_bounds__(256) void gate_pos_kernel(
    const float* __restrict__ g, const float* __restrict__ pos,
    __bf16* __restrict__ gp) {
  const size_t i = (size_t)blockIdx.x * blockDim.x + threadIdx.x;
  float gg = g[i];
  gg = gg > 0.f ? gg : 0.f;
  gp[i] = f2bf(gg * pos[i] + pos[i]);
}

// LSTM cell elementwise: gate order i,f,o (sigmoid) then g (tanh)
__global__ __launch_bounds__(256) void lstm_cell_kernel(
    const float* __restrict__ S, const float* __restrict__ cprev,
    const float* __restrict__ hprev, const float* __restrict__ mask,
    float* __restrict__ hout, float* __restrict__ cout,
    __bf16* __restrict__ hbf, float* __restrict__ hdup) {
  const size_t i = (size_t)blockIdx.x * blockDim.x + threadIdx.x;  // < CB*CR
  const int b = (int)(i >> 10);
  const int r = (int)(i & (CR - 1));
  const float* s = S + (size_t)b * 4 * CR;
  const float ig = 1.f / (1.f + expf(-s[r]));
  const float fg = 1.f / (1.f + expf(-s[CR + r]));
  const float og = 1.f / (1.f + expf(-s[2 * CR + r]));
  const float g  = tanhf(s[3 * CR + r]);
  const float m  = mask[b];
  const float c0 = cprev[i];
  float cn = fg * c0 + ig * g;
  cn = cn * m + c0 * (1.f - m);
  float hn = og * tanhf(cn);
  hn = hn * m + hprev[i] * (1.f - m);
  hout[i] = hn;
  cout[i] = cn;
  if (hbf)  hbf[i] = f2bf(hn);
  if (hdup) hdup[i] = hn;
}

// ---------------------------------------------------------------------------
extern "C" void kernel_launch(void* const* d_in, const int* in_sizes, int n_in,
                              void* d_out, int out_size, void* d_ws, size_t ws_size,
                              hipStream_t stream) {
  (void)in_sizes; (void)n_in; (void)out_size; (void)ws_size;

  const float* xt   = (const float*)d_in[0];
  const float* mask = (const float*)d_in[1];
  const float* V    = (const float*)d_in[2];
  const float* pos  = (const float*)d_in[3];
  const float* h1   = (const float*)d_in[4];
  const float* c1   = (const float*)d_in[5];
  const float* h2   = (const float*)d_in[6];
  const float* c2   = (const float*)d_in[7];
  const float* Wg   = (const float*)d_in[8];
  const float* bg   = (const float*)d_in[9];
  const float* Wi1  = (const float*)d_in[10];
  const float* bi1  = (const float*)d_in[11];
  const float* Wa1  = (const float*)d_in[12];
  const float* ba1  = (const float*)d_in[13];
  const float* Wh1  = (const float*)d_in[14];
  const float* bh1  = (const float*)d_in[15];
  const float* Wi2  = (const float*)d_in[16];
  const float* bi2  = (const float*)d_in[17];
  const float* Wa2  = (const float*)d_in[18];
  const float* ba2  = (const float*)d_in[19];
  const float* Wh2  = (const float*)d_in[20];
  const float* bh2  = (const float*)d_in[21];
  const float* Wv2a = (const float*)d_in[22];
  const float* bv2a = (const float*)d_in[23];
  const float* Wh2a = (const float*)d_in[24];
  const float* bh2a = (const float*)d_in[25];
  const float* Wa2w = (const float*)d_in[26];
  const float* ba2w = (const float*)d_in[27];

  char* wsb = (char*)d_ws;
  size_t off = 0;
  auto alloc = [&](size_t bytes) -> void* {
    void* p = wsb + off;
    off += (bytes + 255) & ~(size_t)255;
    return p;
  };

  // bf16 scratch
  __bf16* Wg_b   = (__bf16*)alloc((size_t)CE * CR * 2);
  __bf16* Wi1_b  = (__bf16*)alloc((size_t)CE * 4 * CR * 2);
  __bf16* Wa1_b  = (__bf16*)alloc((size_t)CR * 4 * CR * 2);
  __bf16* Wh1_b  = (__bf16*)alloc((size_t)CR * 4 * CR * 2);
  __bf16* Wi2_b  = (__bf16*)alloc((size_t)CR * 4 * CR * 2);
  __bf16* Wa2_b  = (__bf16*)alloc((size_t)CR * 4 * CR * 2);
  __bf16* Wh2_b  = (__bf16*)alloc((size_t)CR * 4 * CR * 2);
  __bf16* Wv2a_b = (__bf16*)alloc((size_t)CR * CA * 2);
  __bf16* Wh2a_b = (__bf16*)alloc((size_t)2 * CR * CA * 2);
  __bf16* xt_b   = (__bf16*)alloc((size_t)CB * CE * 2);
  __bf16* V_b    = (__bf16*)alloc((size_t)CB * CL * CR * 2);
  __bf16* h12_b  = (__bf16*)alloc((size_t)CB * 2 * CR * 2);
  __bf16* h1l_b  = (__bf16*)alloc((size_t)CB * CR * 2);
  __bf16* h2l_b  = (__bf16*)alloc((size_t)CB * CR * 2);
  __bf16* gp_b   = (__bf16*)alloc((size_t)CB * CR * 2);
  __bf16* out1_b = (__bf16*)alloc((size_t)CB * CR * 2);
  __bf16* af_b   = (__bf16*)alloc((size_t)CB * CR * 2);
  // f32 scratch
  float* hq32 = (float*)alloc((size_t)CB * CA * 4);
  float* vw32 = (float*)alloc((size_t)CB * CL * CA * 4);
  float* wsm  = (float*)alloc((size_t)CB * CL * 4);
  float* af32 = (float*)alloc((size_t)CB * CR * 4);
  float* g32  = (float*)alloc((size_t)CB * CR * 4);
  float* s1   = (float*)alloc((size_t)CB * 4 * CR * 4);
  float* s2   = (float*)alloc((size_t)CB * 4 * CR * 4);

  auto cvt = [&](const float* s, __bf16* d, size_t n) {
    const int blocks = (int)((n + 255) / 256);
    cvt_bf16_kernel<<<blocks, 256, 0, stream>>>(s, d, n);
  };
  auto gemm = [&](const __bf16* Ab, const __bf16* Bb, float* Cc,
                  const float* bias, int M, int N, int K, int accum) {
    dim3 g(N / BN, M / BM);
    gemm_bf16_wmma<<<g, 256, 0, stream>>>(Ab, Bb, Cc, bias, M, N, K, accum);
  };

  // 1) operand conversion
  cvt(Wg,   Wg_b,   (size_t)CE * CR);
  cvt(Wi1,  Wi1_b,  (size_t)CE * 4 * CR);
  cvt(Wa1,  Wa1_b,  (size_t)CR * 4 * CR);
  cvt(Wh1,  Wh1_b,  (size_t)CR * 4 * CR);
  cvt(Wi2,  Wi2_b,  (size_t)CR * 4 * CR);
  cvt(Wa2,  Wa2_b,  (size_t)CR * 4 * CR);
  cvt(Wh2,  Wh2_b,  (size_t)CR * 4 * CR);
  cvt(Wv2a, Wv2a_b, (size_t)CR * CA);
  cvt(Wh2a, Wh2a_b, (size_t)2 * CR * CA);
  cvt(xt,   xt_b,   (size_t)CB * CE);
  cvt(V,    V_b,    (size_t)CB * CL * CR);
  concat_h_kernel<<<(CB * CR) / 256, 256, 0, stream>>>(h1, h2, h12_b, h1l_b, h2l_b);

  // 2) attention
  gemm(h12_b, Wh2a_b, hq32, bh2a, CB, CA, 2 * CR, 0);           // hq = [h1|h2] @ Wh2a + bh2a
  gemm(V_b,   Wv2a_b, vw32, bv2a, CB * CL, CA, CR, 0);          // vw = V @ Wv2a + bv2a
  attn_score_softmax<<<CB, 256, 0, stream>>>(hq32, vw32, Wa2w, ba2w, wsm);
  attn_apply<<<CB, 256, 0, stream>>>(V, wsm, af32, af_b);

  // 3) gate
  gemm(xt_b, Wg_b, g32, bg, CB, CR, CE, 0);
  gate_pos_kernel<<<(CB * CR) / 256, 256, 0, stream>>>(g32, pos, gp_b);

  // 4) LSTM layer 1: S1 = xt@Wi1 + gp@Wa1 + h1@Wh1 + biases
  gemm(xt_b,  Wi1_b, s1, bi1, CB, 4 * CR, CE, 0);
  gemm(gp_b,  Wa1_b, s1, ba1, CB, 4 * CR, CR, 1);
  gemm(h1l_b, Wh1_b, s1, bh1, CB, 4 * CR, CR, 1);

  float* out  = (float*)d_out;
  float* out2_a = out;
  float* out1_o = out + (size_t)CB * CR;
  float* c1n_o  = out + 2 * (size_t)CB * CR;
  float* out2_b = out + 3 * (size_t)CB * CR;
  float* c2n_o  = out + 4 * (size_t)CB * CR;

  lstm_cell_kernel<<<(CB * CR) / 256, 256, 0, stream>>>(
      s1, c1, h1, mask, out1_o, c1n_o, out1_b, (float*)nullptr);

  // 5) LSTM layer 2: S2 = out1@Wi2 + af@Wa2 + h2@Wh2 + biases
  gemm(out1_b, Wi2_b, s2, bi2, CB, 4 * CR, CR, 0);
  gemm(af_b,   Wa2_b, s2, ba2, CB, 4 * CR, CR, 1);
  gemm(h2l_b,  Wh2_b, s2, bh2, CB, 4 * CR, CR, 1);

  lstm_cell_kernel<<<(CB * CR) / 256, 256, 0, stream>>>(
      s2, c2, h2, mask, out2_b, c2n_o, (__bf16*)nullptr, out2_a);
}
